// BaselineTransformer_24300924960795
// MI455X (gfx1250) — matrix operand order repaired and checked by reading, not compiled, not run on hardware
//
#include <hip/hip_runtime.h>
#include <math.h>

#define D_MODELc 512
#define N_HEADSc 8
#define DKc      64
#define N_LAYERSc 4
#define VOCABc   32000
#define T_SEQc   2048
#define BATCHc   2
#define D_FFc    2048
#define NTOKc    (BATCHc * T_SEQc)

typedef __attribute__((ext_vector_type(16))) __bf16          v16bf;
typedef __attribute__((ext_vector_type(8)))  float           v8f;
typedef __attribute__((ext_vector_type(8)))  unsigned short  v8us;
typedef __attribute__((ext_vector_type(4)))  int             v4i;

union BF16Frag { v16bf v; unsigned short u[16]; v8us h[2]; };

#define AS1 __attribute__((address_space(1)))
#define AS3 __attribute__((address_space(3)))

#if defined(__has_builtin)
#if __has_builtin(__builtin_amdgcn_global_load_async_to_lds_b128)
#define USE_ASYNC_LDS 1
#endif
#endif

#ifdef USE_ASYNC_LDS
// DMA 16 bytes global -> LDS without touching VGPRs (tracked by ASYNCcnt).
__device__ __forceinline__ void async_cp16(void* lds, const void* g) {
  __builtin_amdgcn_global_load_async_to_lds_b128(
      (AS1 v4i*)g, (AS3 v4i*)lds, 0, 0);
}
__device__ __forceinline__ void async_wait() {
  asm volatile("s_wait_asynccnt 0x0" ::: "memory");
}
#endif

__device__ __forceinline__ unsigned short f2bf(float f) {
  union { float f; unsigned int u; } x; x.f = f;
  unsigned int u = x.u;
  u += 0x7fffu + ((u >> 16) & 1u);          // round-to-nearest-even
  return (unsigned short)(u >> 16);
}

// ---------------------------------------------------------------- convert f32 -> bf16
__global__ __launch_bounds__(256) void convert_bf16_kernel(
    const float* __restrict__ in, unsigned short* __restrict__ out, int n) {
  int i = blockIdx.x * 256 + threadIdx.x;
  if (i < n) out[i] = f2bf(in[i]);
}

// ---------------------------------------------------------------- embedding + pos
__global__ __launch_bounds__(128) void embed_kernel(
    const int* __restrict__ x, const float* __restrict__ ew,
    const float* __restrict__ pw, float* __restrict__ h) {
  int tok = blockIdx.x;
  int t   = tok & (T_SEQc - 1);
  int id  = x[tok];
  for (int c = threadIdx.x; c < D_MODELc; c += 128)
    h[(size_t)tok * D_MODELc + c] =
        ew[(size_t)id * D_MODELc + c] + pw[(size_t)t * D_MODELc + c];
}

// ---------------------------------------------------------------- LayerNorm -> bf16 (one wave32 / token)
__global__ __launch_bounds__(32) void layernorm_kernel(
    const float* __restrict__ x, const float* __restrict__ g,
    const float* __restrict__ bta, unsigned short* __restrict__ out) {
  int tok  = blockIdx.x;
  int lane = threadIdx.x;
  const float* xr = x + (size_t)tok * D_MODELc;
  float loc[16];
  float s = 0.0f;
  for (int j = 0; j < 16; ++j) { loc[j] = xr[lane + 32 * j]; s += loc[j]; }
  for (int m = 1; m < 32; m <<= 1) s += __shfl_xor(s, m, 32);
  float mu = s * (1.0f / 512.0f);
  float v = 0.0f;
  for (int j = 0; j < 16; ++j) { float d = loc[j] - mu; v += d * d; }
  for (int m = 1; m < 32; m <<= 1) v += __shfl_xor(v, m, 32);
  float rs = rsqrtf(v * (1.0f / 512.0f) + 1e-5f);
  for (int j = 0; j < 16; ++j) {
    int c = lane + 32 * j;
    out[(size_t)tok * D_MODELc + c] = f2bf((loc[j] - mu) * rs * g[c] + bta[c]);
  }
}

// ---------------------------------------------------------------- bf16 WMMA GEMM
// C[M,N] = act(A[M,K] @ W[N,K]^T + bias);  64x64 block tile, 4 waves, K step 32.
// mode: 0 = store f32 (nontemporal), 1 = accumulate into f32, 2 = store bf16
__global__ __launch_bounds__(128) void gemm_bf16_kernel(
    const unsigned short* __restrict__ A, const unsigned short* __restrict__ W,
    const float* __restrict__ bias, float* __restrict__ Cf,
    unsigned short* __restrict__ Cb, int M, int N, int K, int mode, int doGelu) {
  __shared__ alignas(16) unsigned short sA[64][32];
  __shared__ alignas(16) unsigned short sW[64][32];
  const int tid  = threadIdx.x;
  const int wave = tid >> 5;
  const int lane = tid & 31;
  const int hi   = lane >> 4;
  const int ln   = lane & 15;
  const int r0   = blockIdx.y * 64;
  const int c0   = blockIdx.x * 64;

  v8f acc[4] = {};

  for (int k0 = 0; k0 < K; k0 += 32) {
    // stage 64x32 bf16 tiles of A and W (256 16B chunks each)
#ifdef USE_ASYNC_LDS
    for (int ci = tid; ci < 512; ci += 128) {
      const int c   = ci & 255;
      const int row = c >> 2;       // 0..63
      const int cc  = c & 3;        // 16B chunk within the 32-wide row
      if (ci < 256)
        async_cp16(&sA[row][cc * 8], &A[(size_t)(r0 + row) * K + k0 + cc * 8]);
      else
        async_cp16(&sW[row][cc * 8], &W[(size_t)(c0 + row) * K + k0 + cc * 8]);
    }
    async_wait();
    __syncthreads();
#else
    for (int ci = tid; ci < 512; ci += 128) {
      const int c   = ci & 255;
      const int row = c >> 2;
      const int cc  = c & 3;
      if (ci < 256) {
        v8us t = *(const v8us*)&A[(size_t)(r0 + row) * K + k0 + cc * 8];
        *(v8us*)&sA[row][cc * 8] = t;
        if (k0 + 32 < K)
          __builtin_prefetch(&A[(size_t)(r0 + row) * K + k0 + 32 + cc * 8], 0, 1);
      } else {
        v8us t = *(const v8us*)&W[(size_t)(c0 + row) * K + k0 + cc * 8];
        *(v8us*)&sW[row][cc * 8] = t;
        if (k0 + 32 < K)
          __builtin_prefetch(&W[(size_t)(c0 + row) * K + k0 + 32 + cc * 8], 0, 1);
      }
    }
    __syncthreads();
#endif

    // A fragment: lane holds row (wave*16 + ln); K halves split across lane halves
    BF16Frag a;
    const int mr = wave * 16 + ln;
    a.h[0] = *(const v8us*)&sA[mr][hi * 8];
    a.h[1] = *(const v8us*)&sA[mr][16 + hi * 8];
    for (int nf = 0; nf < 4; ++nf) {
      BF16Frag bfr;
      const int nr = nf * 16 + ln;
      bfr.h[0] = *(const v8us*)&sW[nr][hi * 8];
      bfr.h[1] = *(const v8us*)&sW[nr][16 + hi * 8];
      acc[nf] = __builtin_amdgcn_wmma_f32_16x16x32_bf16(
          false, a.v, false, bfr.v, (short)0, acc[nf], false, false);
    }
    __syncthreads();
  }

  for (int nf = 0; nf < 4; ++nf) {
    const int col = c0 + nf * 16 + ln;
    const float bv = bias ? bias[col] : 0.0f;
    for (int i = 0; i < 8; ++i) {
      float v = acc[nf][i] + bv;
      if (doGelu) v = 0.5f * v * (1.0f + erff(v * 0.70710678118f));
      const int row = r0 + wave * 16 + hi * 8 + i;
      const size_t o = (size_t)row * N + col;
      if (mode == 0)       __builtin_nontemporal_store(v, &Cf[o]);
      else if (mode == 1)  Cf[o] += v;
      else                 Cb[o] = f2bf(v);
    }
  }
}

// ---------------------------------------------------------------- causal flash attention
// One wave per (batch, head, 16-row query tile). dk = 64. QK^T and PV via WMMA.
// K/V 32x64 tiles staged to LDS (async DMA when available).
__global__ __launch_bounds__(32) void attention_kernel(
    const unsigned short* __restrict__ qkv, unsigned short* __restrict__ attn) {
  __shared__ alignas(16) unsigned short sK[32][64];
  __shared__ alignas(16) unsigned short sV[32][64];
  __shared__ alignas(16) unsigned short sP[16][32];
  const int lane = threadIdx.x & 31;
  const int hi   = lane >> 4;
  const int ln   = lane & 15;
  const int qt   = blockIdx.x;   // query tile (16 rows)
  const int head = blockIdx.y;
  const int b    = blockIdx.z;
  const int q0   = qt * 16;

  // Q A-fragments for dk chunks of 32 (128-bit global loads, all 16B aligned)
  BF16Frag qa[2];
  {
    const unsigned short* qrow =
        qkv + (size_t)(b * T_SEQc + q0 + ln) * 1536 + head * DKc;
    for (int c = 0; c < 2; ++c) {
      qa[c].h[0] = *(const v8us*)&qrow[c * 32 + hi * 8];
      qa[c].h[1] = *(const v8us*)&qrow[c * 32 + 16 + hi * 8];
    }
  }

  float mrun[8], lrun[8];
  for (int i = 0; i < 8; ++i) { mrun[i] = -3.0e38f; lrun[i] = 0.0f; }
  v8f O[4] = {};

  const int nkt = qt / 2 + 1;  // 32-key tiles needed for causal coverage
  for (int kt = 0; kt < nkt; ++kt) {
    // stage K and V tiles (32 rows x 64 cols bf16 each = 256 16B chunks each)
    const size_t baseK =
        (size_t)(b * T_SEQc + kt * 32) * 1536 + D_MODELc + head * DKc;
    const size_t baseV = baseK + D_MODELc;
    __syncthreads();
#ifdef USE_ASYNC_LDS
    for (int ci = lane; ci < 512; ci += 32) {
      const int c   = ci & 255;
      const int row = c >> 3;      // 0..31
      const int cc  = c & 7;       // 16B chunk within 64-wide row
      if (ci < 256)
        async_cp16(&sK[row][cc * 8], &qkv[baseK + (size_t)row * 1536 + cc * 8]);
      else
        async_cp16(&sV[row][cc * 8], &qkv[baseV + (size_t)row * 1536 + cc * 8]);
    }
    async_wait();
#else
    for (int ci = lane; ci < 512; ci += 32) {
      const int c   = ci & 255;
      const int row = c >> 3;
      const int cc  = c & 7;
      if (ci < 256) {
        v8us t = *(const v8us*)&qkv[baseK + (size_t)row * 1536 + cc * 8];
        *(v8us*)&sK[row][cc * 8] = t;
      } else {
        v8us t = *(const v8us*)&qkv[baseV + (size_t)row * 1536 + cc * 8];
        *(v8us*)&sV[row][cc * 8] = t;
      }
    }
#endif
    __syncthreads();

    // S tile (16 x 32) as two 16x16 C fragments
    v8f S[2] = {};
    for (int s = 0; s < 2; ++s) {
      for (int c = 0; c < 2; ++c) {
        BF16Frag kf;
        kf.h[0] = *(const v8us*)&sK[s * 16 + ln][c * 32 + hi * 8];
        kf.h[1] = *(const v8us*)&sK[s * 16 + ln][c * 32 + 16 + hi * 8];
        S[s] = __builtin_amdgcn_wmma_f32_16x16x32_bf16(
            false, qa[c].v, false, kf.v, (short)0, S[s], false, false);
      }
    }

    // streaming softmax (rows live as (i, lane-half); columns across 16 lanes)
    float sv[2][8], tmax[8];
    for (int i = 0; i < 8; ++i) {
      const int qpos = q0 + hi * 8 + i;
      float s0 = S[0][i] * 0.125f;
      float s1 = S[1][i] * 0.125f;
      if (kt * 32 + ln > qpos)      s0 = -3.0e38f;
      if (kt * 32 + 16 + ln > qpos) s1 = -3.0e38f;
      sv[0][i] = s0; sv[1][i] = s1;
      tmax[i] = fmaxf(s0, s1);
    }
    for (int i = 0; i < 8; ++i)
      for (int m = 1; m < 16; m <<= 1)
        tmax[i] = fmaxf(tmax[i], __shfl_xor(tmax[i], m, 32));

    float alpha[8], e0[8], e1[8], lsum[8];
    for (int i = 0; i < 8; ++i) {
      const float mnew = fmaxf(mrun[i], tmax[i]);
      alpha[i] = expf(mrun[i] - mnew);
      e0[i] = (sv[0][i] <= -1.0e38f) ? 0.0f : expf(sv[0][i] - mnew);
      e1[i] = (sv[1][i] <= -1.0e38f) ? 0.0f : expf(sv[1][i] - mnew);
      lsum[i] = e0[i] + e1[i];
      mrun[i] = mnew;
    }
    for (int i = 0; i < 8; ++i)
      for (int m = 1; m < 16; m <<= 1)
        lsum[i] += __shfl_xor(lsum[i], m, 32);
    for (int i = 0; i < 8; ++i) lrun[i] = lrun[i] * alpha[i] + lsum[i];
    for (int nf = 0; nf < 4; ++nf)
      for (int i = 0; i < 8; ++i) O[nf][i] *= alpha[i];

    // re-layout P: C-fragment -> A-fragment via LDS
    __syncthreads();
    for (int i = 0; i < 8; ++i) {
      sP[hi * 8 + i][ln]      = f2bf(e0[i]);
      sP[hi * 8 + i][16 + ln] = f2bf(e1[i]);
    }
    __syncthreads();
    BF16Frag pa;
    pa.h[0] = *(const v8us*)&sP[ln][hi * 8];
    pa.h[1] = *(const v8us*)&sP[ln][16 + hi * 8];

    // O += P @ V  (V tile 32 x 64, four 16-wide n-tiles; transposed LDS gather)
    for (int nf = 0; nf < 4; ++nf) {
      BF16Frag vf;
      for (int j = 0; j < 16; ++j)
        vf.u[j] = sV[(j >> 3) * 16 + hi * 8 + (j & 7)][nf * 16 + ln];
      O[nf] = __builtin_amdgcn_wmma_f32_16x16x32_bf16(
          false, pa.v, false, vf.v, (short)0, O[nf], false, false);
    }
  }

  for (int nf = 0; nf < 4; ++nf)
    for (int i = 0; i < 8; ++i) {
      const float v = O[nf][i] / lrun[i];
      const int row = b * T_SEQc + q0 + hi * 8 + i;
      attn[(size_t)row * D_MODELc + head * DKc + nf * 16 + ln] = f2bf(v);
    }
}

// ---------------------------------------------------------------- host orchestration
extern "C" void kernel_launch(void* const* d_in, const int* in_sizes, int n_in,
                              void* d_out, int out_size, void* d_ws, size_t ws_size,
                              hipStream_t stream) {
  (void)in_sizes; (void)n_in; (void)out_size; (void)ws_size;
  const int*   x       = (const int*)d_in[0];
  const float* embed_w = (const float*)d_in[1];
  const float* pos_w   = (const float*)d_in[2];
  const float* ln1_g   = (const float*)d_in[3];
  const float* ln1_b   = (const float*)d_in[4];
  const float* Wqkv    = (const float*)d_in[5];
  const float* Wo      = (const float*)d_in[6];
  const float* ln2_g   = (const float*)d_in[7];
  const float* ln2_b   = (const float*)d_in[8];
  const float* W1      = (const float*)d_in[9];
  const float* b1      = (const float*)d_in[10];
  const float* W2      = (const float*)d_in[11];
  const float* b2      = (const float*)d_in[12];
  const float* lnf_g   = (const float*)d_in[13];
  const float* lnf_b   = (const float*)d_in[14];
  const float* out_w   = (const float*)d_in[15];
  float* logits = (float*)d_out;

  char*  ws  = (char*)d_ws;
  size_t off = 0;
  auto alloc = [&](size_t bytes) -> void* {
    void* p = ws + off;
    off += (bytes + 255) & ~(size_t)255;
    return p;
  };
  float*          h       = (float*)alloc((size_t)NTOKc * D_MODELc * 4);
  unsigned short* lnbuf   = (unsigned short*)alloc((size_t)NTOKc * D_MODELc * 2);
  unsigned short* qkvbuf  = (unsigned short*)alloc((size_t)NTOKc * 3 * D_MODELc * 2);
  unsigned short* attnbuf = (unsigned short*)alloc((size_t)NTOKc * D_MODELc * 2);
  unsigned short* ffbuf   = (unsigned short*)alloc((size_t)NTOKc * D_FFc * 2);
  unsigned short* wqkv_bf = (unsigned short*)alloc((size_t)N_LAYERSc * 3 * D_MODELc * D_MODELc * 2);
  unsigned short* wo_bf   = (unsigned short*)alloc((size_t)N_LAYERSc * D_MODELc * D_MODELc * 2);
  unsigned short* w1_bf   = (unsigned short*)alloc((size_t)N_LAYERSc * D_FFc * D_MODELc * 2);
  unsigned short* w2_bf   = (unsigned short*)alloc((size_t)N_LAYERSc * D_MODELc * D_FFc * 2);
  unsigned short* outw_bf = (unsigned short*)alloc((size_t)VOCABc * D_MODELc * 2);

  auto cvt = [&](const float* src, unsigned short* dst, int n) {
    convert_bf16_kernel<<<dim3((n + 255) / 256), dim3(256), 0, stream>>>(src, dst, n);
  };
  cvt(Wqkv,  wqkv_bf, N_LAYERSc * 3 * D_MODELc * D_MODELc);
  cvt(Wo,    wo_bf,   N_LAYERSc * D_MODELc * D_MODELc);
  cvt(W1,    w1_bf,   N_LAYERSc * D_FFc * D_MODELc);
  cvt(W2,    w2_bf,   N_LAYERSc * D_MODELc * D_FFc);
  cvt(out_w, outw_bf, VOCABc * D_MODELc);

  embed_kernel<<<dim3(NTOKc), dim3(128), 0, stream>>>(x, embed_w, pos_w, h);

  auto gemm = [&](const unsigned short* A, const unsigned short* W,
                  const float* bias, float* Cf, unsigned short* Cb,
                  int M, int N, int K, int mode, int gelu) {
    gemm_bf16_kernel<<<dim3(N / 64, M / 64), dim3(128), 0, stream>>>(
        A, W, bias, Cf, Cb, M, N, K, mode, gelu);
  };

  for (int l = 0; l < N_LAYERSc; ++l) {
    layernorm_kernel<<<dim3(NTOKc), dim3(32), 0, stream>>>(
        h, ln1_g + l * D_MODELc, ln1_b + l * D_MODELc, lnbuf);
    gemm(lnbuf, wqkv_bf + (size_t)l * 3 * D_MODELc * D_MODELc, nullptr,
         nullptr, qkvbuf, NTOKc, 3 * D_MODELc, D_MODELc, 2, 0);
    attention_kernel<<<dim3(T_SEQc / 16, N_HEADSc, BATCHc), dim3(32), 0, stream>>>(
        qkvbuf, attnbuf);
    gemm(attnbuf, wo_bf + (size_t)l * D_MODELc * D_MODELc, nullptr,
         h, nullptr, NTOKc, D_MODELc, D_MODELc, 1, 0);
    layernorm_kernel<<<dim3(NTOKc), dim3(32), 0, stream>>>(
        h, ln2_g + l * D_MODELc, ln2_b + l * D_MODELc, lnbuf);
    gemm(lnbuf, w1_bf + (size_t)l * D_FFc * D_MODELc, b1 + l * D_FFc,
         nullptr, ffbuf, NTOKc, D_FFc, D_MODELc, 2, 1);
    gemm(ffbuf, w2_bf + (size_t)l * D_MODELc * D_FFc, b2 + l * D_MODELc,
         h, nullptr, NTOKc, D_MODELc, D_FFc, 1, 0);
  }

  layernorm_kernel<<<dim3(NTOKc), dim3(32), 0, stream>>>(h, lnf_g, lnf_b, lnbuf);
  gemm(lnbuf, outw_bf, nullptr, logits, nullptr, NTOKc, VOCABc, D_MODELc, 0, 0);
}
